// Net_73263552135202
// MI455X (gfx1250) — matrix operand order repaired and checked by reading, not compiled, not run on hardware
//
#include <hip/hip_runtime.h>
#include <hip/hip_bf16.h>
#include <cstdint>

// ---- Mixtral-mini config (must match reference) ----
#define T_TOK  1024      // B*S tokens
#define DMODEL 512
#define NLAYER 2
#define NHEAD  8
#define NKV    2
#define HDIM   64
#define FFN    1024
#define NEXP   8
#define SEQ    1024
#define VOCAB  32000
#define VCHUNK 4096
#define RMS_EPS 1e-5f

typedef __attribute__((ext_vector_type(16))) __bf16 v16bf;
typedef __attribute__((ext_vector_type(8)))  __bf16 v8bf;
typedef __attribute__((ext_vector_type(4)))  __bf16 v4bf;
typedef __attribute__((ext_vector_type(2)))  __bf16 v2bf;
typedef __attribute__((ext_vector_type(8)))  float  v8f;
typedef __attribute__((ext_vector_type(4)))  float  v4f;

// ============================================================================
// Generic GEMM: C[M,N] (+)= alpha * A[M,K] @ B[K,N]
// A fp32 row-major (lda). B fp32 row-major (ldb), or [N,K] row-major if transB.
// fp32 tiles staged to LDS as bf16, computed with v_wmma_f32_16x16x32_bf16
// (fp32 accumulation). Block = 128 threads (4 wave32) computing a 128x64 C
// tile; each wave owns a 64x32 quadrant as 4x2 WMMA 16x16 accumulators.
// Fast path (full tile in-bounds): b128 global loads + packed bf16 LDS stores
// + global_prefetch of the next K tile. Slow guarded path only for edge tiles.
// ============================================================================
__global__ __launch_bounds__(128) void gemm_bf16_wmma(
    const float* __restrict__ A, int lda,
    const float* __restrict__ B, int ldb, int transB,
    float* __restrict__ C, int ldc,
    int M, int N, int K, int accum, float alpha)
{
  __shared__ alignas(16) __bf16 As[128][32];  // [m][k]
  __shared__ alignas(16) __bf16 Bs[64][32];   // [n][k]
  const int tid  = threadIdx.x;
  const int wave = tid >> 5;
  const int lane = tid & 31;
  const int g    = lane >> 4;   // lane half: 0 or 1
  const int r    = lane & 15;
  const int row0 = blockIdx.y * 128;
  const int col0 = blockIdx.x * 64;
  const int wm   = (wave >> 1) * 64;   // wave's M offset within tile
  const int wn   = (wave & 1)  * 32;   // wave's N offset within tile

  const bool full = (row0 + 128 <= M) && (col0 + 64 <= N) && ((K & 31) == 0);

  v8f vz = {0.f,0.f,0.f,0.f,0.f,0.f,0.f,0.f};
  v8f acc[4][2];
#pragma unroll
  for (int i = 0; i < 4; ++i) { acc[i][0] = vz; acc[i][1] = vz; }

  for (int k0 = 0; k0 < K; k0 += 32) {
    if (full) {
      // ---- A tile: 128x32 fp32, b128 loads, packed bf16 b64 stores ----
#pragma unroll
      for (int ii = 0; ii < 8; ++ii) {
        int idx = tid + 128 * ii;
        int rr = idx >> 3, c4 = (idx & 7) << 2;
        v4f f = *(const v4f*)(A + (size_t)(row0 + rr) * lda + k0 + c4);
        v4bf b = { (__bf16)f[0], (__bf16)f[1], (__bf16)f[2], (__bf16)f[3] };
        *(v4bf*)&As[rr][c4] = b;
      }
      if (transB) {
        // B is [N,K] row-major: same pattern as A
#pragma unroll
        for (int ii = 0; ii < 4; ++ii) {
          int idx = tid + 128 * ii;
          int nn = idx >> 3, c4 = (idx & 7) << 2;
          v4f f = *(const v4f*)(B + (size_t)(col0 + nn) * ldb + k0 + c4);
          v4bf b = { (__bf16)f[0], (__bf16)f[1], (__bf16)f[2], (__bf16)f[3] };
          *(v4bf*)&Bs[nn][c4] = b;
        }
      } else {
        // B is [K,N] row-major: load 2 adjacent k-rows (coalesced along n),
        // pack (k,k+1) bf16 pairs -> 32-bit LDS stores into Bs[n][k].
#pragma unroll
        for (int ii = 0; ii < 2; ++ii) {
          int idx = tid + 128 * ii;
          int kk = (idx >> 4) << 1;         // even k in 0..30
          int n4 = (idx & 15) << 2;         // n in 0..60 step 4
          v4f f0 = *(const v4f*)(B + (size_t)(k0 + kk)     * ldb + col0 + n4);
          v4f f1 = *(const v4f*)(B + (size_t)(k0 + kk + 1) * ldb + col0 + n4);
#pragma unroll
          for (int j = 0; j < 4; ++j) {
            v2bf p = { (__bf16)f0[j], (__bf16)f1[j] };
            *(v2bf*)&Bs[n4 + j][kk] = p;
          }
        }
      }
      // prefetch next K tile into cache (global_prefetch_b8)
      if (k0 + 32 < K) {
        __builtin_prefetch(A + (size_t)(row0 + (tid & 127)) * lda + k0 + 32, 0, 1);
        if (transB)
          __builtin_prefetch(B + (size_t)(col0 + (tid & 63)) * ldb + k0 + 32, 0, 1);
        else if (k0 + 64 <= K)
          __builtin_prefetch(B + (size_t)(k0 + 32 + (tid & 31)) * ldb + col0, 0, 1);
      }
    } else {
      // ---- guarded scalar staging for edge tiles (e.g. router N=8) ----
      for (int i = tid; i < 128 * 32; i += 128) {
        int rr = i >> 5, cc = i & 31;
        int gr = row0 + rr, gc = k0 + cc;
        float v = (gr < M && gc < K) ? A[(size_t)gr * lda + gc] : 0.f;
        As[rr][cc] = (__bf16)v;
      }
      for (int i = tid; i < 64 * 32; i += 128) {
        int nn = i >> 5, kk = i & 31;
        int gn = col0 + nn, gk = k0 + kk;
        float v = 0.f;
        if (gn < N && gk < K)
          v = transB ? B[(size_t)gn * ldb + gk] : B[(size_t)gk * ldb + gn];
        Bs[nn][kk] = (__bf16)v;
      }
    }
    __syncthreads();

    // ---- compute: cache both B fragments, stream 4 A fragments ----
    v16bf bfr[2];
#pragma unroll
    for (int j = 0; j < 2; ++j) {
      // B fragment (32x16 bf16): lane-half g holds k = 16*(e/8) + 8g + (e%8)
      const __bf16* brow = &Bs[wn + 16*j + r][0];
      v8bf blo = *(const v8bf*)(brow + 8*g);
      v8bf bhi = *(const v8bf*)(brow + 16 + 8*g);
      bfr[j] = __builtin_shufflevector(blo, bhi,
          0,1,2,3,4,5,6,7,8,9,10,11,12,13,14,15);
    }
#pragma unroll
    for (int i = 0; i < 4; ++i) {
      // A fragment (16x32 bf16): lane-half g holds k = 16*(e/8) + 8g + (e%8)
      const __bf16* arow = &As[wm + 16*i + r][0];
      v8bf alo = *(const v8bf*)(arow + 8*g);
      v8bf ahi = *(const v8bf*)(arow + 16 + 8*g);
      v16bf af = __builtin_shufflevector(alo, ahi,
          0,1,2,3,4,5,6,7,8,9,10,11,12,13,14,15);
      acc[i][0] = __builtin_amdgcn_wmma_f32_16x16x32_bf16(
          false, af, false, bfr[0], (short)0, acc[i][0], false, false);
      acc[i][1] = __builtin_amdgcn_wmma_f32_16x16x32_bf16(
          false, af, false, bfr[1], (short)0, acc[i][1], false, false);
    }
    __syncthreads();
  }

  // C/D layout: VGPR e -> row = base + 8*g + e, col = n0 + r
#pragma unroll
  for (int i = 0; i < 4; ++i)
#pragma unroll
    for (int j = 0; j < 2; ++j) {
      int cidx = col0 + wn + 16*j + r;
      if (cidx >= N) continue;
      int rbase = row0 + wm + 16*i + 8*g;
#pragma unroll
      for (int e = 0; e < 8; ++e) {
        int rr = rbase + e;
        if (rr < M) {
          size_t o = (size_t)rr * ldc + cidx;
          float v = alpha * acc[i][j][e];
          C[o] = accum ? (C[o] + v) : v;
        }
      }
    }
}

// ============================================================================
// Elementwise / reduction kernels
// ============================================================================
__global__ void embed_kernel(const int* __restrict__ ids,
                             const float* __restrict__ emb,
                             float* __restrict__ x)
{
  int idx = blockIdx.x * blockDim.x + threadIdx.x;
  if (idx >= T_TOK * DMODEL) return;
  int t = idx / DMODEL, d = idx % DMODEL;
  x[idx] = emb[(size_t)ids[t] * DMODEL + d];
}

__global__ void rmsnorm_kernel(const float* __restrict__ x,
                               const float* __restrict__ w,
                               float* __restrict__ out)
{
  int t = blockIdx.x;
  const float* xr = x + (size_t)t * DMODEL;
  __shared__ float red[256];
  int tid = threadIdx.x;
  float s = 0.f;
  for (int i = tid; i < DMODEL; i += 256) { float v = xr[i]; s += v * v; }
  red[tid] = s; __syncthreads();
  for (int st = 128; st > 0; st >>= 1) {
    if (tid < st) red[tid] += red[tid + st];
    __syncthreads();
  }
  float inv = rsqrtf(red[0] / (float)DMODEL + RMS_EPS);
  for (int i = tid; i < DMODEL; i += 256)
    out[(size_t)t * DMODEL + i] = xr[i] * inv * w[i];
}

__global__ void rope_kernel(float* __restrict__ p, int nh)
{
  int idx = blockIdx.x * blockDim.x + threadIdx.x;
  int total = T_TOK * nh * (HDIM / 2);
  if (idx >= total) return;
  int half = idx & 31;
  int hd   = (idx >> 5) % nh;
  int t    = idx / (32 * nh);
  float inv = powf(1.0e6f, -(float)half / 32.0f);
  float ang = (float)t * inv;
  float c = cosf(ang), s = sinf(ang);
  float* base = p + (size_t)t * nh * HDIM + hd * HDIM;
  float x0 = base[half], x1 = base[half + 32];
  base[half]      = x0 * c - x1 * s;
  base[half + 32] = x1 * c + x0 * s;
}

__global__ void softmax_causal_kernel(float* __restrict__ sc)
{
  int qp = blockIdx.x, hh = blockIdx.y;
  float* row = sc + ((size_t)hh * SEQ + qp) * SEQ;
  int n = qp + 1;                         // causal: keys 0..qp
  const float scale = 0.125f;             // 1/sqrt(64)
  __shared__ float red[256];
  int tid = threadIdx.x;

  float m = -1e30f;
  for (int i = tid; i < n; i += 256) m = fmaxf(m, row[i] * scale);
  red[tid] = m; __syncthreads();
  for (int st = 128; st > 0; st >>= 1) {
    if (tid < st) red[tid] = fmaxf(red[tid], red[tid + st]);
    __syncthreads();
  }
  m = red[0]; __syncthreads();

  float ssum = 0.f;
  for (int i = tid; i < n; i += 256) {
    float e = __expf(row[i] * scale - m);
    row[i] = e; ssum += e;
  }
  red[tid] = ssum; __syncthreads();
  for (int st = 128; st > 0; st >>= 1) {
    if (tid < st) red[tid] += red[tid + st];
    __syncthreads();
  }
  float inv = 1.f / red[0];
  for (int i = tid; i < n; i += 256) row[i] *= inv;
  for (int i = n + tid; i < SEQ; i += 256) row[i] = 0.f;
}

__global__ void router_topk_kernel(const float* __restrict__ rtr,
                                   float* __restrict__ we)
{
  int t = blockIdx.x * blockDim.x + threadIdx.x;
  if (t >= T_TOK) return;
  float p[NEXP];
  float m = -1e30f;
  for (int e = 0; e < NEXP; ++e) { p[e] = rtr[t * NEXP + e]; m = fmaxf(m, p[e]); }
  float sum = 0.f;
  for (int e = 0; e < NEXP; ++e) { p[e] = __expf(p[e] - m); sum += p[e]; }
  for (int e = 0; e < NEXP; ++e) p[e] /= sum;
  int i1 = 0;
  for (int e = 1; e < NEXP; ++e) if (p[e] > p[i1]) i1 = e;      // first max
  int i2 = (i1 == 0) ? 1 : 0;
  for (int e = 0; e < NEXP; ++e)
    if (e != i1 && p[e] > p[i2]) i2 = e;
  float norm = p[i1] + p[i2];
  for (int e = 0; e < NEXP; ++e)
    we[t * NEXP + e] = (e == i1 ? p[i1] : (e == i2 ? p[i2] : 0.f)) / norm;
}

__global__ void silu_mul_kernel(float* __restrict__ a,
                                const float* __restrict__ b)
{
  int idx = blockIdx.x * blockDim.x + threadIdx.x;
  if (idx >= T_TOK * FFN) return;
  float av = a[idx];
  float g = av / (1.f + __expf(-av));     // silu
  a[idx] = g * b[idx];
}

__global__ void moe_accum_kernel(float* __restrict__ x,
                                 const float* __restrict__ y,
                                 const float* __restrict__ we, int e)
{
  int idx = blockIdx.x * blockDim.x + threadIdx.x;
  if (idx >= T_TOK * DMODEL) return;
  int t = idx / DMODEL;
  x[idx] += we[t * NEXP + e] * y[idx];
}

// Online log-sum-exp over vocab chunks + label-logit gather.
__global__ void lse_update_kernel(const float* __restrict__ lg, int cn, int c0,
                                  int first, const int* __restrict__ labels,
                                  float* __restrict__ mrun,
                                  float* __restrict__ srun,
                                  float* __restrict__ labv)
{
  int t = blockIdx.x;
  const float* row = lg + (size_t)t * cn;
  __shared__ float red[256];
  int tid = threadIdx.x;

  float m = -1e30f;
  for (int i = tid; i < cn; i += 256) m = fmaxf(m, row[i]);
  red[tid] = m; __syncthreads();
  for (int st = 128; st > 0; st >>= 1) {
    if (tid < st) red[tid] = fmaxf(red[tid], red[tid + st]);
    __syncthreads();
  }
  float mc = red[0];
  float mprev = first ? -1e30f : mrun[t];
  float newm = fmaxf(mprev, mc);
  __syncthreads();

  float s = 0.f;
  for (int i = tid; i < cn; i += 256) s += __expf(row[i] - newm);
  red[tid] = s; __syncthreads();
  for (int st = 128; st > 0; st >>= 1) {
    if (tid < st) red[tid] += red[tid + st];
    __syncthreads();
  }
  if (tid == 0) {
    float sprev = first ? 0.f : srun[t];
    srun[t] = sprev * __expf(mprev - newm) + red[0];
    mrun[t] = newm;
    int L = labels[t];
    if (L >= c0 && L < c0 + cn) labv[t] = row[L - c0];
  }
}

__global__ void loss_kernel(const float* __restrict__ mrun,
                            const float* __restrict__ srun,
                            const float* __restrict__ labv,
                            float* __restrict__ out)
{
  __shared__ float red[256];
  int tid = threadIdx.x;
  float s = 0.f;
  for (int t = tid; t < T_TOK; t += 256)
    s += (mrun[t] + logf(srun[t]) - labv[t]);   // -logp(label)
  red[tid] = s; __syncthreads();
  for (int st = 128; st > 0; st >>= 1) {
    if (tid < st) red[tid] += red[tid + st];
    __syncthreads();
  }
  if (tid == 0) out[0] = red[0] / (float)T_TOK;
}

// ============================================================================
// Host orchestration
// ============================================================================
extern "C" void kernel_launch(void* const* d_in, const int* in_sizes, int n_in,
                              void* d_out, int out_size, void* d_ws, size_t ws_size,
                              hipStream_t stream)
{
  (void)in_sizes; (void)n_in; (void)out_size; (void)ws_size;
  // d_in order (dict insertion order):
  // 0:input_ids 1:labels 2:embed
  // per layer l (base=3+10l): ln1 ln2 wq wk wv wo wg w1 w2 w3
  // 3+10L: fnorm, 3+10L+1: lm_head
  const int*   input_ids = (const int*)d_in[0];
  const int*   labels    = (const int*)d_in[1];
  const float* embedW    = (const float*)d_in[2];
  const float* fnorm     = (const float*)d_in[3 + NLAYER * 10];
  const float* lm_head   = (const float*)d_in[3 + NLAYER * 10 + 1];

  float* ws = (float*)d_ws;
  size_t off = 0;
  auto alloc = [&](size_t n) { float* p = ws + off; off += n; return p; };
  float* x      = alloc((size_t)T_TOK * DMODEL);
  float* h      = alloc((size_t)T_TOK * DMODEL);
  float* q      = alloc((size_t)T_TOK * NHEAD * HDIM);
  float* kbuf   = alloc((size_t)T_TOK * NKV * HDIM);
  float* vbuf   = alloc((size_t)T_TOK * NKV * HDIM);
  float* attno  = alloc((size_t)T_TOK * NHEAD * HDIM);
  float* scores = alloc((size_t)NHEAD * SEQ * SEQ);
  float* rtr    = alloc((size_t)T_TOK * NEXP);
  float* wecomb = alloc((size_t)T_TOK * NEXP);
  float* ea     = alloc((size_t)T_TOK * FFN);
  float* eb     = alloc((size_t)T_TOK * FFN);
  float* ey     = alloc((size_t)T_TOK * DMODEL);
  float* lg     = alloc((size_t)T_TOK * VCHUNK);
  float* mrun   = alloc(T_TOK);
  float* srun   = alloc(T_TOK);
  float* labv   = alloc(T_TOK);

  auto gemm = [&](const float* Ap, int lda, const float* Bp, int ldb, int tB,
                  float* Cp, int ldc, int M, int N, int K, int accum) {
    dim3 grid((unsigned)((N + 63) / 64), (unsigned)((M + 127) / 128));
    gemm_bf16_wmma<<<grid, 128, 0, stream>>>(Ap, lda, Bp, ldb, tB,
                                             Cp, ldc, M, N, K, accum, 1.0f);
  };

  embed_kernel<<<(T_TOK * DMODEL + 255) / 256, 256, 0, stream>>>(input_ids, embedW, x);

  for (int l = 0; l < NLAYER; ++l) {
    int base = 3 + l * 10;
    const float* ln1 = (const float*)d_in[base + 0];
    const float* ln2 = (const float*)d_in[base + 1];
    const float* wq  = (const float*)d_in[base + 2];
    const float* wk  = (const float*)d_in[base + 3];
    const float* wv  = (const float*)d_in[base + 4];
    const float* wo  = (const float*)d_in[base + 5];
    const float* wg  = (const float*)d_in[base + 6];
    const float* w1  = (const float*)d_in[base + 7];
    const float* w2  = (const float*)d_in[base + 8];
    const float* w3  = (const float*)d_in[base + 9];

    // ---- attention ----
    rmsnorm_kernel<<<T_TOK, 256, 0, stream>>>(x, ln1, h);
    gemm(h, DMODEL, wq, NHEAD * HDIM, 0, q,    NHEAD * HDIM, T_TOK, NHEAD * HDIM, DMODEL, 0);
    gemm(h, DMODEL, wk, NKV * HDIM,  0, kbuf, NKV * HDIM,   T_TOK, NKV * HDIM,   DMODEL, 0);
    gemm(h, DMODEL, wv, NKV * HDIM,  0, vbuf, NKV * HDIM,   T_TOK, NKV * HDIM,   DMODEL, 0);
    rope_kernel<<<(T_TOK * NHEAD * (HDIM / 2) + 255) / 256, 256, 0, stream>>>(q, NHEAD);
    rope_kernel<<<(T_TOK * NKV   * (HDIM / 2) + 255) / 256, 256, 0, stream>>>(kbuf, NKV);

    for (int hh = 0; hh < NHEAD; ++hh) {
      int kvh = hh / (NHEAD / NKV);
      // scores_h = Q_h @ K_h^T   (transB: K stored [S, kv*HD])
      gemm(q + hh * HDIM, NHEAD * HDIM, kbuf + kvh * HDIM, NKV * HDIM, 1,
           scores + (size_t)hh * SEQ * SEQ, SEQ, SEQ, SEQ, HDIM, 0);
    }
    softmax_causal_kernel<<<dim3(SEQ, NHEAD), 256, 0, stream>>>(scores);
    for (int hh = 0; hh < NHEAD; ++hh) {
      int kvh = hh / (NHEAD / NKV);
      // O_h = P_h @ V_h
      gemm(scores + (size_t)hh * SEQ * SEQ, SEQ, vbuf + kvh * HDIM, NKV * HDIM, 0,
           attno + hh * HDIM, NHEAD * HDIM, SEQ, HDIM, SEQ, 0);
    }
    // x += O @ Wo   (accumulate = residual add)
    gemm(attno, NHEAD * HDIM, wo, DMODEL, 0, x, DMODEL, T_TOK, DMODEL, NHEAD * HDIM, 1);

    // ---- MoE ----
    rmsnorm_kernel<<<T_TOK, 256, 0, stream>>>(x, ln2, h);
    gemm(h, DMODEL, wg, NEXP, 0, rtr, NEXP, T_TOK, NEXP, DMODEL, 0);
    router_topk_kernel<<<(T_TOK + 255) / 256, 256, 0, stream>>>(rtr, wecomb);
    for (int e = 0; e < NEXP; ++e) {
      const float* w1e = w1 + (size_t)e * DMODEL * FFN;
      const float* w3e = w3 + (size_t)e * DMODEL * FFN;
      const float* w2e = w2 + (size_t)e * FFN * DMODEL;
      gemm(h, DMODEL, w1e, FFN, 0, ea, FFN, T_TOK, FFN, DMODEL, 0);
      gemm(h, DMODEL, w3e, FFN, 0, eb, FFN, T_TOK, FFN, DMODEL, 0);
      silu_mul_kernel<<<(T_TOK * FFN + 255) / 256, 256, 0, stream>>>(ea, eb);
      gemm(ea, FFN, w2e, DMODEL, 0, ey, DMODEL, T_TOK, DMODEL, FFN, 0);
      moe_accum_kernel<<<(T_TOK * DMODEL + 255) / 256, 256, 0, stream>>>(x, ey, wecomb, e);
    }
  }

  // ---- final norm + chunked lm_head + streaming log-softmax loss ----
  rmsnorm_kernel<<<T_TOK, 256, 0, stream>>>(x, fnorm, h);
  for (int c = 0; c < VOCAB; c += VCHUNK) {
    int cn = (VOCAB - c < VCHUNK) ? (VOCAB - c) : VCHUNK;
    gemm(h, DMODEL, lm_head + c, VOCAB, 0, lg, cn, T_TOK, cn, DMODEL, 0);
    lse_update_kernel<<<T_TOK, 256, 0, stream>>>(lg, cn, c, (c == 0) ? 1 : 0,
                                                 labels, mrun, srun, labv);
  }
  loss_kernel<<<1, 256, 0, stream>>>(mrun, srun, labv, (float*)d_out);
}
